// MHGCDA_56727928045715
// MI455X (gfx1250) — matrix operand order repaired and checked by reading, not compiled, not run on hardware
//
#include <hip/hip_runtime.h>
#include <math.h>

// ---------------- problem constants (match reference) ----------------
#define NNODES 10000
#define NEDGES 170000        // E + N self loops
#define HEADS  8
#define DHID   64
#define FIN    4096
#define FHID   512           // HEADS*DHID
#define NCLS   5
#define HC     40            // HEADS*NCLS
#define SLOPE  0.2f

typedef __attribute__((ext_vector_type(16))) __bf16 v16bf;
typedef __attribute__((ext_vector_type(8)))  __bf16 v8bf;
typedef __attribute__((ext_vector_type(8)))  float  v8f;

// LDS layout: B tile stored column-major, [col][k], stride padded to 40 bf16
// (80 B = 5*16 B): each col's 16-K fragment read is a 16B-aligned pair of
// ds_load_b128; lane c starts at dword c*20 mod 64 -> all 16 lanes' b128
// reads tile the 64 banks conflict-free.
#define BSTRIDE 40

// ---------------- GEMM: C = A(MxK) * B(KxN), fp32 in/out, bf16x3 WMMA ----
// Block = 8 waves = 8 M-tiles x one 64-col group. B k-tile staged via LDS.
// Fragment layouts per ISA 7.12.2:
//   A (16-bit 16x32): lane l: row=l&15, kbase=(l>>4)*8, elem i -> K=kbase+(i<8?i:i+8)
//   B (16-bit 32x16): lane l: col=l&15, kbase=(l>>4)*8, elem i -> K=kbase+(i<8?i:i+8)
//   C/D (32-bit 16x16): lane l: col=l&15, VGPR r -> row=r+(l>>4)*8
__global__ __launch_bounds__(256)
void gemm_bf16x3_kernel(const float* __restrict__ A, const float* __restrict__ B,
                        float* __restrict__ C, int M, int K, int Ncols,
                        int mtiles, int ngroups)
{
    __shared__ __align__(16) __bf16 s_bh[64 * BSTRIDE];
    __shared__ __align__(16) __bf16 s_bl[64 * BSTRIDE];

    const int lane = threadIdx.x & 31;
    const int wave = threadIdx.x >> 5;
    const int mb   = blockIdx.x / ngroups;
    const int ng   = blockIdx.x % ngroups;
    const int mt   = mb * 8 + wave;
    const bool mvalid = (mt < mtiles);
    const int mtc  = mvalid ? mt : (mtiles - 1);   // clamped: all waves join barriers

    const int r16  = lane & 15;
    const int hi16 = lane >> 4;
    int arow = mtc * 16 + r16;
    if (arow >= M) arow = M - 1;                   // safe load; store is guarded

    // staging coords: 256 threads cover 32 k-rows x 64 cols, 8 cols/thread
    const int skk = threadIdx.x >> 3;              // k row 0..31
    const int scb = (threadIdx.x & 7) * 8;         // col base 0..56
    const int sgc = ng * 64 + scb;                 // global col base
    const bool aligned64 = ((Ncols & 63) == 0);

    v8f acc[4];
#pragma unroll
    for (int t = 0; t < 4; ++t)
#pragma unroll
        for (int r = 0; r < 8; ++r) acc[t][r] = 0.0f;

    for (int k0 = 0; k0 < K; k0 += 32) {
        // ---- stage B k-tile into LDS as bf16 hi/lo, column-major ----
        __syncthreads();                            // prev compute done
        {
            float xv[8];
            const size_t rowoff = (size_t)(k0 + skk) * Ncols;
            if (aligned64) {
                const float4 x0 = *(const float4*)(B + rowoff + sgc);
                const float4 x1 = *(const float4*)(B + rowoff + sgc + 4);
                xv[0] = x0.x; xv[1] = x0.y; xv[2] = x0.z; xv[3] = x0.w;
                xv[4] = x1.x; xv[5] = x1.y; xv[6] = x1.z; xv[7] = x1.w;
            } else {
#pragma unroll
                for (int c = 0; c < 8; ++c) {
                    int col = sgc + c;
                    int colc = (col < Ncols) ? col : (Ncols - 1);
                    float x = B[rowoff + colc];
                    xv[c] = (col < Ncols) ? x : 0.0f;
                }
            }
#pragma unroll
            for (int c = 0; c < 8; ++c) {
                __bf16 h = (__bf16)xv[c];
                s_bh[(scb + c) * BSTRIDE + skk] = h;
                s_bl[(scb + c) * BSTRIDE + skk] = (__bf16)(xv[c] - (float)h);
            }
        }
        __syncthreads();                            // staging visible

        // ---- A fragment: 4 x b128, split into bf16 hi/lo ----
        const float4* ap4 = (const float4*)(A + (size_t)arow * K + k0 + hi16 * 8);
        const float4 a0 = ap4[0], a1 = ap4[1], a2 = ap4[4], a3 = ap4[5];
        float af[16];
        af[0]=a0.x; af[1]=a0.y; af[2]=a0.z; af[3]=a0.w;
        af[4]=a1.x; af[5]=a1.y; af[6]=a1.z; af[7]=a1.w;
        af[8]=a2.x; af[9]=a2.y; af[10]=a2.z; af[11]=a2.w;
        af[12]=a3.x; af[13]=a3.y; af[14]=a3.z; af[15]=a3.w;
        v16bf ah, alo;
#pragma unroll
        for (int i = 0; i < 16; ++i) {
            __bf16 h = (__bf16)af[i];
            ah[i]  = h;
            alo[i] = (__bf16)(af[i] - (float)h);
        }

        // ---- 4 n-tiles: B fragments from LDS, 3 WMMAs each (bf16x3) ----
#pragma unroll
        for (int t = 0; t < 4; ++t) {
            const int col = t * 16 + r16;
            const __bf16* ph = s_bh + col * BSTRIDE + hi16 * 8;
            const __bf16* pl = s_bl + col * BSTRIDE + hi16 * 8;
            const v8bf bh0 = *(const v8bf*)(ph);
            const v8bf bh1 = *(const v8bf*)(ph + 16);
            const v8bf bl0 = *(const v8bf*)(pl);
            const v8bf bl1 = *(const v8bf*)(pl + 16);
            v16bf bh, bl;
#pragma unroll
            for (int i = 0; i < 8; ++i) {
                bh[i] = bh0[i]; bh[i + 8] = bh1[i];
                bl[i] = bl0[i]; bl[i + 8] = bl1[i];
            }
            acc[t] = __builtin_amdgcn_wmma_f32_16x16x32_bf16(false, ah,  false, bh,
                                                             (short)0, acc[t], false, false);
            acc[t] = __builtin_amdgcn_wmma_f32_16x16x32_bf16(false, ah,  false, bl,
                                                             (short)0, acc[t], false, false);
            acc[t] = __builtin_amdgcn_wmma_f32_16x16x32_bf16(false, alo, false, bh,
                                                             (short)0, acc[t], false, false);
        }
    }

    // ---- store D (guarded; padding waves/cols drop out here) ----
    if (mvalid) {
#pragma unroll
        for (int t = 0; t < 4; ++t) {
            const int col = ng * 64 + t * 16 + r16;
            if (col < Ncols) {
#pragma unroll
                for (int r = 0; r < 8; ++r) {
                    const int row = mt * 16 + hi16 * 8 + r;
                    if (row < M) C[(size_t)row * Ncols + col] = acc[t][r];
                }
            }
        }
    }
}

// ---------------- GAT edge-softmax stages ----------------
__device__ __forceinline__ void atomicMaxF32(float* addr, float val) {
    if (val >= 0.0f) atomicMax((int*)addr, __float_as_int(val));
    else             atomicMin((unsigned int*)addr, __float_as_uint(val));
}

// per (node, head): el/er dot products; init emax=-inf, denom=0
__global__ void node_prep_kernel(const float* __restrict__ feat,
                                 const float* __restrict__ al,
                                 const float* __restrict__ ar,
                                 float* el, float* er, float* emax, float* denom,
                                 int D)
{
    int idx = blockIdx.x * blockDim.x + threadIdx.x;     // node*HEADS + h
    if (idx >= NNODES * HEADS) return;
    int h = idx & (HEADS - 1);
    const float* f = feat + (size_t)idx * D;
    float sl = 0.0f, sr = 0.0f;
    for (int d = 0; d < D; ++d) { float v = f[d]; sl += v * al[h * D + d]; sr += v * ar[h * D + d]; }
    el[idx] = sl; er[idx] = sr;
    emax[idx]  = -INFINITY;
    denom[idx] = 0.0f;
}

__global__ void fill_kernel(float* __restrict__ p, float v, size_t n)
{
    size_t i = (size_t)blockIdx.x * blockDim.x + threadIdx.x;
    if (i < n) p[i] = v;
}

// per (edge, head): e = leaky_relu(el[src]+er[dst]); segment-max into emax[dst]
__global__ void edge_logit_kernel(const int* __restrict__ src, const int* __restrict__ dst,
                                  const float* __restrict__ el, const float* __restrict__ er,
                                  float* __restrict__ eedge, float* __restrict__ emax, int ne)
{
    int idx = blockIdx.x * blockDim.x + threadIdx.x;     // e*HEADS + h
    if (idx >= ne * HEADS) return;
    int e = idx >> 3, h = idx & 7;
    int s = src[e], d = dst[e];
    float x = el[s * HEADS + h] + er[d * HEADS + h];
    x = (x > 0.0f) ? x : SLOPE * x;
    eedge[idx] = x;
    atomicMaxF32(&emax[d * HEADS + h], x);
}

// per (edge, head): ex = exp(e - emax[dst]); denom[dst] += ex
__global__ void edge_exp_kernel(const int* __restrict__ dst, const float* __restrict__ emax,
                                float* __restrict__ eedge, float* __restrict__ denom, int ne)
{
    int idx = blockIdx.x * blockDim.x + threadIdx.x;
    if (idx >= ne * HEADS) return;
    int e = idx >> 3, h = idx & 7;
    int d = dst[e];
    float ex = expf(eedge[idx] - emax[d * HEADS + h]);
    eedge[idx] = ex;
    atomicAdd(&denom[d * HEADS + h], ex);
}

// per (edge, head): out[dst,h,:] += (ex/denom[dst,h]) * feat[src,h,:]
__global__ void edge_scatter_kernel(const int* __restrict__ src, const int* __restrict__ dst,
                                    const float* __restrict__ eedge, const float* __restrict__ denom,
                                    const float* __restrict__ feat, float* __restrict__ out,
                                    int ne, int D)
{
    int idx = blockIdx.x * blockDim.x + threadIdx.x;
    if (idx >= ne * HEADS) return;
    int e = idx >> 3, h = idx & 7;
    int s = src[e], d = dst[e];
    float alpha = eedge[idx] / denom[d * HEADS + h];
    const float* fs = feat + ((size_t)s * HEADS + h) * D;
    float*       od = out  + ((size_t)d * HEADS + h) * D;
    for (int dd = 0; dd < D; ++dd) atomicAdd(&od[dd], alpha * fs[dd]);
}

// h = elu(out + b)   over N*F elements, bias repeats every F
__global__ void elu_bias_kernel(const float* __restrict__ out, const float* __restrict__ b,
                                float* __restrict__ dstp, int total, int F)
{
    int i = blockIdx.x * blockDim.x + threadIdx.x;
    if (i >= total) return;
    float x = out[i] + b[i % F];
    dstp[i] = (x > 0.0f) ? x : expm1f(x);
}

// pred[n,c] = mean_h( out[n,h,c] + res[n,h,c] + bc[h,c] )
__global__ void cls_epilogue_kernel(const float* __restrict__ out, const float* __restrict__ res,
                                    const float* __restrict__ bc, float* __restrict__ pred)
{
    int idx = blockIdx.x * blockDim.x + threadIdx.x;     // node*NCLS + c
    if (idx >= NNODES * NCLS) return;
    int node = idx / NCLS, c = idx % NCLS;
    float s = 0.0f;
    for (int h = 0; h < HEADS; ++h) {
        int j = (node * HEADS + h) * NCLS + c;
        s += out[j] + res[j] + bc[h * NCLS + c];
    }
    pred[idx] = s * (1.0f / HEADS);
}

// ---------------- host-side helpers ----------------
static void run_gemm(const float* A, const float* B, float* C,
                     int M, int K, int Nc, hipStream_t s)
{
    int mtiles  = (M + 15) / 16;
    int ngroups = (Nc + 63) / 64;
    int mblocks = (mtiles + 7) / 8;
    gemm_bf16x3_kernel<<<mblocks * ngroups, 256, 0, s>>>(A, B, C, M, K, Nc, mtiles, ngroups);
}

static void run_gat_edges(const float* feat, const float* al, const float* ar,
                          const int* src, const int* dst, int D, int ne,
                          float* el, float* er, float* emax, float* denom, float* eedge,
                          float* outbuf, hipStream_t s)
{
    int nh = NNODES * HEADS;
    node_prep_kernel<<<(nh + 255) / 256, 256, 0, s>>>(feat, al, ar, el, er, emax, denom, D);
    size_t outn = (size_t)NNODES * HEADS * D;
    fill_kernel<<<(unsigned)((outn + 255) / 256), 256, 0, s>>>(outbuf, 0.0f, outn);
    int eh = ne * HEADS;
    edge_logit_kernel<<<(eh + 255) / 256, 256, 0, s>>>(src, dst, el, er, eedge, emax, ne);
    edge_exp_kernel<<<(eh + 255) / 256, 256, 0, s>>>(dst, emax, eedge, denom, ne);
    edge_scatter_kernel<<<(eh + 255) / 256, 256, 0, s>>>(src, dst, eedge, denom, feat, outbuf, ne, D);
}

extern "C" void kernel_launch(void* const* d_in, const int* in_sizes, int n_in,
                              void* d_out, int out_size, void* d_ws, size_t ws_size,
                              hipStream_t stream)
{
    (void)n_in; (void)out_size; (void)ws_size;
    // -------- inputs (setup_inputs dict order) --------
    const int*   SRC[3] = { (const int*)d_in[0], (const int*)d_in[2], (const int*)d_in[4] };
    const int*   DST[3] = { (const int*)d_in[1], (const int*)d_in[3], (const int*)d_in[5] };
    const float* X[3]   = { (const float*)d_in[6], (const float*)d_in[7], (const float*)d_in[8] };
    const float* W0  = (const float*)d_in[9];
    const float* al0 = (const float*)d_in[10];
    const float* ar0 = (const float*)d_in[11];
    const float* b0  = (const float*)d_in[12];
    const float* W1  = (const float*)d_in[13];
    const float* al1 = (const float*)d_in[14];
    const float* ar1 = (const float*)d_in[15];
    const float* b1  = (const float*)d_in[16];
    const float* Wc[2]  = { (const float*)d_in[17], (const float*)d_in[22] };
    const float* alc[2] = { (const float*)d_in[18], (const float*)d_in[23] };
    const float* arc[2] = { (const float*)d_in[19], (const float*)d_in[24] };
    const float* bc[2]  = { (const float*)d_in[20], (const float*)d_in[25] };
    const float* rW[2]  = { (const float*)d_in[21], (const float*)d_in[26] };
    const int ne = in_sizes[0];   // 170000

    // -------- workspace layout (floats) --------
    float* ws     = (float*)d_ws;
    float* featA  = ws;                                   // N*FHID
    float* h0buf  = featA  + (size_t)NNODES * FHID;       // N*FHID (also residual buf)
    float* outbuf = h0buf  + (size_t)NNODES * FHID;       // N*FHID
    float* elbuf  = outbuf + (size_t)NNODES * FHID;       // N*H
    float* erbuf  = elbuf  + (size_t)NNODES * HEADS;
    float* emaxb  = erbuf  + (size_t)NNODES * HEADS;
    float* denomb = emaxb  + (size_t)NNODES * HEADS;
    float* eedge  = denomb + (size_t)NNODES * HEADS;      // NE*H

    // -------- outputs --------
    float* out = (float*)d_out;
    float* pred[4] = { out, out + NNODES * NCLS, out + 2 * NNODES * NCLS, out + 3 * NNODES * NCLS };
    float* emb[3];
    emb[0] = out + 4 * NNODES * NCLS;
    emb[1] = emb[0] + (size_t)NNODES * FHID;
    emb[2] = emb[1] + (size_t)NNODES * FHID;

    const int nf = NNODES * FHID;

    // -------- shared-weight encoder on all 3 graphs --------
    for (int g = 0; g < 3; ++g) {
        // layer 0
        run_gemm(X[g], W0, featA, NNODES, FIN, FHID, stream);
        run_gat_edges(featA, al0, ar0, SRC[g], DST[g], DHID, ne,
                      elbuf, erbuf, emaxb, denomb, eedge, outbuf, stream);
        elu_bias_kernel<<<(nf + 255) / 256, 256, 0, stream>>>(outbuf, b0, h0buf, nf, FHID);
        // layer 1  -> embedding written straight into d_out
        run_gemm(h0buf, W1, featA, NNODES, FHID, FHID, stream);
        run_gat_edges(featA, al1, ar1, SRC[g], DST[g], DHID, ne,
                      elbuf, erbuf, emaxb, denomb, eedge, outbuf, stream);
        elu_bias_kernel<<<(nf + 255) / 256, 256, 0, stream>>>(outbuf, b1, emb[g], nf, FHID);
    }

    // -------- classifiers: (graph, classifier) -> pred slot --------
    const int jobs[4][2] = { {0, 0}, {1, 1}, {2, 0}, {2, 1} };
    for (int j = 0; j < 4; ++j) {
        const int g = jobs[j][0], c = jobs[j][1];
        run_gemm(emb[g], Wc[c], featA, NNODES, FHID, HC, stream);   // attention features
        run_gat_edges(featA, alc[c], arc[c], SRC[g], DST[g], NCLS, ne,
                      elbuf, erbuf, emaxb, denomb, eedge, outbuf, stream);
        run_gemm(emb[g], rW[c], h0buf, NNODES, FHID, HC, stream);   // residual projection
        int nc = NNODES * NCLS;
        cls_epilogue_kernel<<<(nc + 255) / 256, 256, 0, stream>>>(outbuf, h0buf, bc[c], pred[j]);
    }
}
